// VectorQuantizer_90787018703005
// MI455X (gfx1250) — matrix operand order repaired and checked by reading, not compiled, not run on hardware
//
#include <hip/hip_runtime.h>

// Problem constants (from reference): N=32768, K=8192, D=256, BETA=1.0
#define NN 32768
#define KK 8192
#define DD 256
#define KSPLIT 4     // blocks splitting the K dimension (merged via u64 atomicMax)
#define WAVES 8      // waves per block (256 threads, wave32)
#define ROWT 2       // 16-row tiles per wave (A stays in registers, B shared)
#define KC 32        // codes per TDM chunk staged in LDS
#define PITCH 264    // LDS row pitch in bf16 elems (512B data + 16B pad -> bank rotate)

typedef __attribute__((ext_vector_type(8)))  __bf16 v8bf;
typedef __attribute__((ext_vector_type(16))) __bf16 v16bf;
typedef __attribute__((ext_vector_type(8)))  float  v8f;
typedef unsigned int u32x4 __attribute__((ext_vector_type(4)));
typedef int          i32x8 __attribute__((ext_vector_type(8)));
typedef int          i32x4 __attribute__((ext_vector_type(4)));

#if __has_include(<hip/amd_detail/amd_gfx1250_TDM.h>)
// amdgpu-toolchain (clang-23): 6-arg builtin
#define TDM_LOAD(g0, g1, g2, g3, gz) __builtin_amdgcn_tensor_load_to_lds(g0, g1, g2, g3, gz, 0)
#else
// ROCm 7.2 (clang-22): 5-arg builtin
#define TDM_LOAD(g0, g1, g2, g3, gz) __builtin_amdgcn_tensor_load_to_lds(g0, g1, g2, g3, 0)
#endif

// Issue one TDM load: KC x 256 bf16 tile (rows of emb) -> LDS with 16B/row padding.
__device__ __forceinline__ void tdm_load_chunk(const __bf16* gsrc, unsigned lds_off) {
  unsigned long long ga = (unsigned long long)(uintptr_t)gsrc;
  u32x4 g0;
  g0.x = 1u;                                               // count=1 (valid), user mode
  g0.y = lds_off;                                          // LDS byte address
  g0.z = (unsigned)(ga & 0xFFFFFFFFu);                     // global_addr[31:0]
  g0.w = (unsigned)((ga >> 32) & 0x1FFFFFFu) | (2u << 30); // global_addr[56:32], type=2
  i32x8 g1;
  g1[0] = (1 << 16)        // data_size = 2 bytes
        | (1 << 20)        // pad_enable
        | (6 << 22)        // pad_interval: 128 dwords (= one 512B row)
        | (3 << 25);       // pad_amount: 4 dwords (16B) -> pitch 528B
  g1[1] = (int)(256u << 16);   // tensor_dim0 = 256 (elems)
  g1[2] = (int)(8192u << 16);  // tensor_dim0 hi=0 ; tensor_dim1 = 8192
  g1[3] = (int)(256u << 16);   // tensor_dim1 hi=0 ; tile_dim0 = 256
  g1[4] = KC;                  // tile_dim1 = KC rows, tile_dim2 = 0
  g1[5] = 256;                 // tensor_dim0_stride = 256 (elems)
  g1[6] = 0;
  g1[7] = 0;
  i32x4 z4 = {0, 0, 0, 0};
  i32x8 z8 = {0, 0, 0, 0, 0, 0, 0, 0};
  (void)z8;
  TDM_LOAD(g0, g1, z4, z4, z8);
}

// ---------------- convert z -> bf16 ----------------
__global__ __launch_bounds__(256) void vq_convert_z(const float* __restrict__ z,
                                                    __bf16* __restrict__ zb) {
  int i = (blockIdx.x * 256 + threadIdx.x) * 4;
#pragma unroll
  for (int u = 0; u < 4; ++u) zb[i + u] = (__bf16)z[i + u];
}

// ---------------- convert emb -> bf16 + row norms ----------------
__global__ __launch_bounds__(256) void vq_convert_e(const float* __restrict__ emb,
                                                    __bf16* __restrict__ eb,
                                                    float* __restrict__ en) {
  __shared__ float red[WAVES];
  int k = blockIdx.x;
  int d = threadIdx.x;
  float f = emb[(size_t)k * DD + d];
  eb[(size_t)k * DD + d] = (__bf16)f;
  float sq = f * f;
#pragma unroll
  for (int off = 16; off >= 1; off >>= 1) sq += __shfl_down(sq, off);
  int lane = threadIdx.x & 31, wv = threadIdx.x >> 5;
  if (lane == 0) red[wv] = sq;
  __syncthreads();
  if (threadIdx.x == 0) {
    float t = 0.f;
#pragma unroll
    for (int i = 0; i < WAVES; ++i) t += red[i];
    en[k] = t;
  }
}

// ---------------- fused GEMM (bf16 WMMA) + argmin, TDM-staged B ----------------
// score s = 2*(z . e_k) - ||e_k||^2 ; argmax_k s == argmin_k ||z - e_k||^2
__global__ __launch_bounds__(256) void vq_argmin(const __bf16* __restrict__ zb,
                                                 const __bf16* __restrict__ eb,
                                                 const float* __restrict__ en,
                                                 unsigned long long* __restrict__ best) {
  __shared__ __bf16 lds_b[2][KC * PITCH];   // 2 x 16.5 KB double buffer

  const int tid  = threadIdx.x;
  const int lane = tid & 31;
  const int wv   = tid >> 5;
  const int r    = lane & 15;   // row (A) / column-code (B,C) selector
  const int h16  = lane >> 4;   // lane half
  const int rowBase = blockIdx.x * (WAVES * ROWT * 16) + wv * (ROWT * 16);
  const int k0 = blockIdx.y * (KK / KSPLIT);
  const int ITER = (KK / KSPLIT) / KC;

  // A fragments: 16x32 bf16 tiles. Lane half 0: K=d0+0..7 (v0-3), K=d0+16..23 (v4-7);
  // half 1: K=d0+8..15, K=d0+24..31  -> two contiguous 16B loads per lane.
  v16bf a[ROWT][8];
#pragma unroll
  for (int t = 0; t < ROWT; ++t) {
    const __bf16* zr = zb + (size_t)(rowBase + t * 16 + r) * DD;
#pragma unroll
    for (int d = 0; d < 8; ++d) {
      union { v16bf v; v8bf h[2]; } u;
      u.h[0] = *(const v8bf*)(zr + d * 32 + h16 * 8);
      u.h[1] = *(const v8bf*)(zr + d * 32 + 16 + h16 * 8);
      a[t][d] = u.v;
    }
  }

  float bval[ROWT][8];
  int   bk[ROWT][8];
#pragma unroll
  for (int t = 0; t < ROWT; ++t)
#pragma unroll
    for (int j = 0; j < 8; ++j) { bval[t][j] = -3.4e38f; bk[t][j] = 0; }

  // Prologue: wave 0 kicks off chunk 0.
  if (wv == 0)
    tdm_load_chunk(eb + (size_t)k0 * DD, (unsigned)(uintptr_t)&lds_b[0][0]);

  for (int it = 0; it < ITER; ++it) {
    __syncthreads();   // everyone done reading buf[(it+1)&1] from iteration it-1
    if (wv == 0) {
      if (it + 1 < ITER) {
        tdm_load_chunk(eb + (size_t)(k0 + (it + 1) * KC) * DD,
                       (unsigned)(uintptr_t)&lds_b[(it + 1) & 1][0]);
        __builtin_amdgcn_s_wait_tensorcnt(1);  // chunk `it` complete (in-order TDM)
      } else {
        __builtin_amdgcn_s_wait_tensorcnt(0);
      }
    }
    __syncthreads();   // buf[it&1] now holds chunk `it` for all waves
    const __bf16* bp = &lds_b[it & 1][0];
    const int kb = k0 + it * KC;

#pragma unroll
    for (int ks = 0; ks < KC / 16; ++ks) {
      // B fragment from LDS: lane = code column; lanes 0-15 hold K=0..7,16..23,
      // lanes 16-31 hold K=8..15,24..31 (two 16B ds loads, bank-rotated by the pad).
      const __bf16* bcol = bp + (ks * 16 + r) * PITCH + h16 * 8;
      v8f c[ROWT] = {};
      union BU { v16bf v; v8bf h[2]; };
      BU b0, b1;
      b0.h[0] = *(const v8bf*)(bcol);
      b0.h[1] = *(const v8bf*)(bcol + 16);
#pragma unroll
      for (int d = 0; d < 8; ++d) {
        BU& cur = (d & 1) ? b1 : b0;
        BU& nxt = (d & 1) ? b0 : b1;
        if (d < 7) {   // prefetch next d-chunk while WMMA consumes current
          nxt.h[0] = *(const v8bf*)(bcol + (d + 1) * 32);
          nxt.h[1] = *(const v8bf*)(bcol + (d + 1) * 32 + 16);
        }
#pragma unroll
        for (int t = 0; t < ROWT; ++t)
          c[t] = __builtin_amdgcn_wmma_f32_16x16x32_bf16(
              false, a[t][d], false, cur.v, (short)0, c[t], false, false);
      }
      float env = en[kb + ks * 16 + r];
#pragma unroll
      for (int t = 0; t < ROWT; ++t)
#pragma unroll
        for (int j = 0; j < 8; ++j) {
          float s = 2.0f * c[t][j] - env;       // C/D: row = j + 8*h16, code = kb+ks*16+r
          if (s > bval[t][j]) { bval[t][j] = s; bk[t][j] = kb + ks * 16; }
        }
    }
  }

  // Reduce (val, idx) across the 16 lanes of each half (xor masks 1..8 stay in-half).
#pragma unroll
  for (int t = 0; t < ROWT; ++t)
#pragma unroll
    for (int j = 0; j < 8; ++j) {
      float v = bval[t][j];
      int   idx = bk[t][j] + r;
#pragma unroll
      for (int off = 8; off >= 1; off >>= 1) {
        float ov = __shfl_xor(v, off);
        int   oi = __shfl_xor(idx, off);
        if (ov > v || (ov == v && oi < idx)) { v = ov; idx = oi; }
      }
      if (r == 0) {
        unsigned u = __float_as_uint(v);
        u = (u & 0x80000000u) ? ~u : (u | 0x80000000u);   // order-preserving map
        unsigned long long key =
            ((unsigned long long)u << 32) | (unsigned)(0xFFFFFFFFu - (unsigned)idx);
        int row = rowBase + t * 16 + 8 * h16 + j;
        atomicMax(&best[row], key);
      }
    }
}

// ---------------- gather z_q, loss, idx ----------------
__global__ __launch_bounds__(256) void vq_gather_loss(const float* __restrict__ z,
                                                      const float* __restrict__ emb,
                                                      const unsigned long long* __restrict__ best,
                                                      float* __restrict__ out) {
  __shared__ float red[WAVES];
  int n = blockIdx.x;
  int d = threadIdx.x;
  unsigned long long key = best[n];
  int idx = (int)(0xFFFFFFFFu - (unsigned)(key & 0xFFFFFFFFull));
  float zq = emb[(size_t)idx * DD + d];
  float zv = z[(size_t)n * DD + d];
  out[1 + (size_t)n * DD + d] = zq;               // z_q_st == z_q numerically
  float df = zq - zv;
  float sq = df * df;
#pragma unroll
  for (int off = 16; off >= 1; off >>= 1) sq += __shfl_down(sq, off);
  int lane = threadIdx.x & 31, wv = threadIdx.x >> 5;
  if (lane == 0) red[wv] = sq;
  __syncthreads();
  if (threadIdx.x == 0) {
    float t = 0.f;
#pragma unroll
    for (int i = 0; i < WAVES; ++i) t += red[i];
    // loss = mean((zq-z)^2) + BETA*mean((zq-z)^2) = 2 * sum / (N*D)
    atomicAdd(&out[0], 2.0f * t * (1.0f / ((float)NN * (float)DD)));
    out[1 + (size_t)NN * DD + n] = (float)idx;    // idx output (as float)
  }
}

extern "C" void kernel_launch(void* const* d_in, const int* in_sizes, int n_in,
                              void* d_out, int out_size, void* d_ws, size_t ws_size,
                              hipStream_t stream) {
  const float* z   = (const float*)d_in[0];   // [N, D] f32
  const float* emb = (const float*)d_in[1];   // [K, D] f32
  float* out = (float*)d_out;                 // [0]=loss, [1..N*D]=z_q_st, then idx

  char* w = (char*)d_ws;
  __bf16* zb = (__bf16*)w;                                          // 16 MB
  __bf16* eb = (__bf16*)(w + (size_t)NN * DD * 2);                  //  4 MB
  float*  en = (float*)(w + (size_t)NN * DD * 2 + (size_t)KK * DD * 2);
  unsigned long long* best =
      (unsigned long long*)(w + (size_t)NN * DD * 2 + (size_t)KK * DD * 2 + (size_t)KK * 4);

  hipMemsetAsync(best, 0, (size_t)NN * 8, stream);   // key 0 < any real encoded score
  hipMemsetAsync(out, 0, sizeof(float), stream);     // zero loss accumulator

  vq_convert_z<<<(NN * DD) / 1024, 256, 0, stream>>>(z, zb);
  vq_convert_e<<<KK, 256, 0, stream>>>(emb, eb, en);
  vq_argmin<<<dim3(NN / (WAVES * ROWT * 16), KSPLIT), 256, 0, stream>>>(zb, eb, en, best);
  vq_gather_loss<<<NN, 256, 0, stream>>>(z, emb, best, out);
}
